// MHPooling_90529320666025
// MI455X (gfx1250) — compile-verified, matrix operand order checked by reading, not run on hardware
//
#include <hip/hip_runtime.h>
#include <hip/hip_bf16.h>

typedef __attribute__((ext_vector_type(16))) _Float16 v16h;
typedef __attribute__((ext_vector_type(8)))  float    v8f;

// ---------------- problem constants ----------------
constexpr int BB = 4, S = 1024, D = 1024, H = 16, DK = 64;
constexpr int MROWS = BB * S;          // 4096
constexpr long XE = (long)MROWS * D;   // 4M elements
constexpr long WE = (long)D * D;       // 1M elements

// ---------------- WMMA wrapper ----------------
__device__ __forceinline__ v8f wmma_f16(v16h a, v16h b, v8f c) {
  return __builtin_amdgcn_wmma_f32_16x16x32_f16(
      /*neg_a=*/false, a, /*neg_b=*/false, b,
      /*c_mod=*/(short)0, c, /*reuse_a=*/false, /*reuse_b=*/false);
}

// ------------- CDNA5 async memory->LDS copy (ASYNCcnt tracked) -------------
// global_load_async_to_lds_b128: VDST = LDS byte offset (generic-ptr low 32
// bits, per the LDS aperture rule), VADDR = 64-bit global addr.
__device__ __forceinline__ void async_ld_b128(const _Float16* lds_dst,
                                              const _Float16* gsrc) {
  unsigned lo = (unsigned)(unsigned long long)lds_dst;
  unsigned long long ga = (unsigned long long)gsrc;
  asm volatile("global_load_async_to_lds_b128 %0, %1, off"
               :: "v"(lo), "v"(ga) : "memory");
}
__device__ __forceinline__ void wait_async0() {
  asm volatile("s_wait_asynccnt 0" ::: "memory");
}

// Load a 16x32 f16 fragment from an LDS tile stored k-contiguous per row.
// A-matrix layout (ISA 7.12.2): lane m = lane&15; kbase = 8*(lane>=16);
// VGPR0-3 hold k = kbase..kbase+7, VGPR4-7 hold k = 16+kbase..16+kbase+7.
__device__ __forceinline__ v16h ld_frag(const _Float16* tile, int rbase,
                                        int stride, int k0, int lane) {
  const int m  = lane & 15;
  const int kb = (lane >> 4) << 3;   // 0 or 8
  const _Float16* p = tile + (rbase + m) * stride + k0 + kb;
  union { uint4 u[2]; v16h h; } r;
  r.u[0] = *reinterpret_cast<const uint4*>(p);
  r.u[1] = *reinterpret_cast<const uint4*>(p + 16);
  return r.h;
}

// Row reductions across the 16 lanes of each wave32 half (matches the C/D
// layout: a row lives in lanes 0-15 or 16-31 of one accumulator VGPR).
__device__ __forceinline__ float rowmax16(float x) {
  x = fmaxf(x, __shfl_xor(x, 1, 32));
  x = fmaxf(x, __shfl_xor(x, 2, 32));
  x = fmaxf(x, __shfl_xor(x, 4, 32));
  x = fmaxf(x, __shfl_xor(x, 8, 32));
  return x;
}
__device__ __forceinline__ float rowsum16(float x) {
  x += __shfl_xor(x, 1, 32);
  x += __shfl_xor(x, 2, 32);
  x += __shfl_xor(x, 4, 32);
  x += __shfl_xor(x, 8, 32);
  return x;
}

// ---------------- fp32 -> fp16 convert (straight) ----------------
__global__ void f32_to_f16_kernel(const float* __restrict__ src,
                                  _Float16* __restrict__ dst, int n) {
  for (int i = blockIdx.x * blockDim.x + threadIdx.x; i < n;
       i += gridDim.x * blockDim.x)
    dst[i] = (_Float16)src[i];
}

// -------- fp32 -> fp16 convert + transpose: dst[c][r] = src[r][c] ----------
// Produces weights in [N][K] row-major so GEMM B-fragments are k-contiguous.
__global__ __launch_bounds__(256) void transpose_f32_to_f16_kernel(
    const float* __restrict__ src, _Float16* __restrict__ dst, int n) {
  __shared__ float tile[32][33];
  const int bx = blockIdx.x * 32, by = blockIdx.y * 32;
  const int tx = threadIdx.x & 31, ty = threadIdx.x >> 5;  // 8 rows / pass
#pragma unroll
  for (int i = 0; i < 32; i += 8)
    tile[ty + i][tx] = src[(long)(by + ty + i) * n + bx + tx];
  __syncthreads();
#pragma unroll
  for (int i = 0; i < 32; i += 8)
    dst[(long)(bx + ty + i) * n + by + tx] = (_Float16)tile[tx][ty + i];
}

// ---------------- GEMM: C[MxN] = A[MxK](f16) @ BT[NxK](f16)^T + bias -------
// Block tile 128x128, BK=64, 256 threads = 8 waves; wave w owns rows
// [16w,16w+16) x 128 cols = 8 f32 accumulators; 16 WMMAs per wave per k-step.
// OMODE: 0 = f32 row-major, 1 = f16 row-major, 2 = f16 [B,H,DK,S] (for V).
template <int OMODE>
__global__ __launch_bounds__(256) void gemm_bias_kernel(
    const _Float16* __restrict__ A, const _Float16* __restrict__ BT,
    const float* __restrict__ bias, float* __restrict__ Cf,
    _Float16* __restrict__ Ch, int M, int N, int K) {
  constexpr int BM = 128, BN = 128, BK = 64;
  __shared__ _Float16 a_sh[BM * BK];   // [row][k]
  __shared__ _Float16 b_sh[BN * BK];   // [n][k]

  const int tid = threadIdx.x;
  const int lane = tid & 31, w = tid >> 5;
  const int n0 = blockIdx.x * BN, m0 = blockIdx.y * BM;

  v8f acc[8];
#pragma unroll
  for (int j = 0; j < 8; ++j) acc[j] = {};

  for (int kk = 0; kk < K; kk += BK) {
    __syncthreads();
    // async-stage both tiles: 128 rows x 64 f16 = 8 x b128 per row each
#pragma unroll
    for (int i = 0; i < 4; ++i) {
      int idx = tid + i * 256;            // 0..1023
      int row = idx >> 3, g = (idx & 7) * 8;
      async_ld_b128(a_sh + row * BK + g, A + (long)(m0 + row) * K + kk + g);
      async_ld_b128(b_sh + row * BK + g, BT + (long)(n0 + row) * K + kk + g);
    }
    wait_async0();
    __syncthreads();

    v16h a0 = ld_frag(a_sh, w * 16, BK, 0, lane);
    v16h a1 = ld_frag(a_sh, w * 16, BK, 32, lane);
#pragma unroll
    for (int j = 0; j < 8; ++j) {
      acc[j] = wmma_f16(a0, ld_frag(b_sh, j * 16, BK, 0, lane), acc[j]);
      acc[j] = wmma_f16(a1, ld_frag(b_sh, j * 16, BK, 32, lane), acc[j]);
    }
  }

  // epilogue: C layout -> n = lane&15, m = r + 8*(lane>=16)
  const int nlo = lane & 15, hi = (lane >> 4) & 1;
#pragma unroll
  for (int j = 0; j < 8; ++j) {
    int col = n0 + j * 16 + nlo;
    float bv = bias[col];
#pragma unroll
    for (int r = 0; r < 8; ++r) {
      long row = m0 + w * 16 + r + 8 * hi;
      float val = acc[j][r] + bv;
      if (OMODE == 0) {
        Cf[row * N + col] = val;
      } else if (OMODE == 1) {
        Ch[row * N + col] = (_Float16)val;
      } else {
        // V output in [B, H, DK, S]: row = b*S+s, col = h*DK+dk
        int b = (int)(row >> 10), s = (int)(row & (S - 1));
        int h = col >> 6, dk = col & (DK - 1);
        Ch[(long)((b * H + h) * DK + dk) * S + s] = (_Float16)val;
      }
    }
  }
}

// ---------------- flash attention -------------------------------------
// grid = B*H*(S/64) blocks, 128 threads (4 waves); wave owns 16 query rows.
// Q,K in [B,S,D] (head-sliced); V pre-transposed in [B,H,DK,S].
__global__ __launch_bounds__(128) void attn_kernel(
    const _Float16* __restrict__ Q, const _Float16* __restrict__ Km,
    const _Float16* __restrict__ Vt, _Float16* __restrict__ ctx) {
  constexpr int BQ = 64, BC = 32;
  __shared__ _Float16 q_sh[BQ * DK];      // [q][dk]
  __shared__ _Float16 k_sh[BC * DK];      // [key][dk]  (= K^T in frag layout)
  __shared__ _Float16 vT[DK * BC];        // [dk][key]
  __shared__ _Float16 p_sh[4 * 16 * BC];  // per-wave 16x32 P staging

  const int tid = threadIdx.x, lane = tid & 31, w = tid >> 5;
  const int nsb = S / BQ;                 // 16
  const int sblk = blockIdx.x % nsb;
  const int bh = blockIdx.x / nsb;
  const int b = bh / H, h = bh % H;
  const int q0 = sblk * BQ;
  const int hi = (lane >> 4) & 1, nlo = lane & 15;

  // async-stage Q tile 64x64: 8 x b128 per row, 4 per thread
#pragma unroll
  for (int i = 0; i < 4; ++i) {
    int idx = tid + i * 128;              // 0..511
    int row = idx >> 3, g = (idx & 7) * 8;
    async_ld_b128(q_sh + row * DK + g,
                  Q + (long)(b * S + q0 + row) * D + h * DK + g);
  }
  wait_async0();
  __syncthreads();
  v16h aq0 = ld_frag(q_sh, w * 16, DK, 0, lane);
  v16h aq1 = ld_frag(q_sh, w * 16, DK, 32, lane);

  float mrun[8], lrun[8];
  v8f o0 = {}, o1 = {}, o2 = {}, o3 = {};
#pragma unroll
  for (int r = 0; r < 8; ++r) { mrun[r] = -3.0e38f; lrun[r] = 0.f; }

  _Float16* pw = p_sh + w * 16 * BC;
  const _Float16* vt_base = Vt + (long)bh * DK * S;

  for (int t0 = 0; t0 < S; t0 += BC) {
    __syncthreads();
    // K chunk 32x64: 256 b128, 2/thread ; V^T chunk 64x32: 256 b128, 2/thread
#pragma unroll
    for (int i = 0; i < 2; ++i) {
      int idx = tid + i * 128;            // 0..255
      int row = idx >> 3, g = (idx & 7) * 8;
      async_ld_b128(k_sh + row * DK + g,
                    Km + (long)(b * S + t0 + row) * D + h * DK + g);
      int dk = idx >> 2, gg = (idx & 3) * 8;
      async_ld_b128(vT + dk * BC + gg, vt_base + (long)dk * S + t0 + gg);
    }
    wait_async0();
    __syncthreads();

    // scores: 16 q-rows x 32 keys, contraction over DK=64
    v8f sa = {}, sb = {};
    sa = wmma_f16(aq0, ld_frag(k_sh, 0, DK, 0, lane), sa);
    sa = wmma_f16(aq1, ld_frag(k_sh, 0, DK, 32, lane), sa);
    sb = wmma_f16(aq0, ld_frag(k_sh, 16, DK, 0, lane), sb);
    sb = wmma_f16(aq1, ld_frag(k_sh, 16, DK, 32, lane), sb);

    // online softmax (scale = 1/sqrt(DK) = 0.125)
#pragma unroll
    for (int r = 0; r < 8; ++r) {
      float x0 = sa[r] * 0.125f, x1 = sb[r] * 0.125f;
      float mx = rowmax16(fmaxf(x0, x1));
      float nm = fmaxf(mrun[r], mx);
      float al = __expf(mrun[r] - nm);
      mrun[r] = nm;
      x0 = __expf(x0 - nm);
      x1 = __expf(x1 - nm);
      sa[r] = x0;
      sb[r] = x1;
      lrun[r] = lrun[r] * al + rowsum16(x0 + x1);
      o0[r] *= al; o1[r] *= al; o2[r] *= al; o3[r] *= al;
    }

    // re-fragment P via per-wave LDS staging (C layout -> A layout)
#pragma unroll
    for (int r = 0; r < 8; ++r) {
      pw[(r + 8 * hi) * BC + nlo]      = (_Float16)sa[r];
      pw[(r + 8 * hi) * BC + 16 + nlo] = (_Float16)sb[r];
    }
    v16h ap = ld_frag(pw, 0, BC, 0, lane);
    o0 = wmma_f16(ap, ld_frag(vT, 0,  BC, 0, lane), o0);
    o1 = wmma_f16(ap, ld_frag(vT, 16, BC, 0, lane), o1);
    o2 = wmma_f16(ap, ld_frag(vT, 32, BC, 0, lane), o2);
    o3 = wmma_f16(ap, ld_frag(vT, 48, BC, 0, lane), o3);
  }

  // epilogue: normalize and scatter back to [B,S,D] (heads interleaved)
#pragma unroll
  for (int r = 0; r < 8; ++r) {
    int srow = q0 + w * 16 + r + 8 * hi;
    float inv = 1.0f / lrun[r];
    long base = (long)(b * S + srow) * D + h * DK;
    ctx[base + nlo]      = (_Float16)(o0[r] * inv);
    ctx[base + 16 + nlo] = (_Float16)(o1[r] * inv);
    ctx[base + 32 + nlo] = (_Float16)(o2[r] * inv);
    ctx[base + 48 + nlo] = (_Float16)(o3[r] * inv);
  }
}

// ---------------- host launcher ----------------
extern "C" void kernel_launch(void* const* d_in, const int* in_sizes, int n_in,
                              void* d_out, int out_size, void* d_ws,
                              size_t ws_size, hipStream_t stream) {
  const float* x  = (const float*)d_in[0];
  const float* Wq = (const float*)d_in[1];
  const float* bq = (const float*)d_in[2];
  const float* Wk = (const float*)d_in[3];
  const float* bk = (const float*)d_in[4];
  const float* Wv = (const float*)d_in[5];
  const float* bv = (const float*)d_in[6];
  const float* Wo = (const float*)d_in[7];
  const float* bo = (const float*)d_in[8];
  float* out = (float*)d_out;

  // workspace carve-up (f16): x, 4 weights(T), Q, K, Vt, ctx => 48MB
  _Float16* p = (_Float16*)d_ws;
  _Float16* xh   = p; p += XE;
  _Float16* wqt  = p; p += WE;
  _Float16* wkt  = p; p += WE;
  _Float16* wvt  = p; p += WE;
  _Float16* wot  = p; p += WE;
  _Float16* qh   = p; p += XE;
  _Float16* kh   = p; p += XE;
  _Float16* vt   = p; p += XE;   // [B,H,DK,S]
  _Float16* ctxh = p; p += XE;
  if (ws_size < (size_t)((char*)p - (char*)d_ws)) return;

  // 1) conversions: x straight; weights converted + transposed to [N][K]
  f32_to_f16_kernel<<<2048, 256, 0, stream>>>(x, xh, (int)XE);
  dim3 tgrid(D / 32, D / 32);
  transpose_f32_to_f16_kernel<<<tgrid, 256, 0, stream>>>(Wq, wqt, D);
  transpose_f32_to_f16_kernel<<<tgrid, 256, 0, stream>>>(Wk, wkt, D);
  transpose_f32_to_f16_kernel<<<tgrid, 256, 0, stream>>>(Wv, wvt, D);
  transpose_f32_to_f16_kernel<<<tgrid, 256, 0, stream>>>(Wo, wot, D);

  // 2) Q/K/V projections
  dim3 ggrid(D / 128, MROWS / 128);  // (8, 32)
  gemm_bias_kernel<1><<<ggrid, 256, 0, stream>>>(xh, wqt, bq, nullptr, qh,
                                                 MROWS, D, D);
  gemm_bias_kernel<1><<<ggrid, 256, 0, stream>>>(xh, wkt, bk, nullptr, kh,
                                                 MROWS, D, D);
  gemm_bias_kernel<2><<<ggrid, 256, 0, stream>>>(xh, wvt, bv, nullptr, vt,
                                                 MROWS, D, D);

  // 3) attention: B*H*(S/64) = 1024 blocks of 128 threads
  attn_kernel<<<BB * H * (S / 64), 128, 0, stream>>>(qh, kh, vt, ctxh);

  // 4) output projection (f32 out into d_out)
  gemm_bias_kernel<0><<<ggrid, 256, 0, stream>>>(ctxh, wot, bo, out,
                                                 nullptr, MROWS, D, D);
}
